// SelfAttentionLayer_90950227460615
// MI455X (gfx1250) — matrix operand order repaired and checked by reading, not compiled, not run on hardware
//
#include <hip/hip_runtime.h>
#include <hip/hip_bf16.h>
#include <math.h>

typedef __attribute__((ext_vector_type(16))) __bf16 v16bf;
typedef __attribute__((ext_vector_type(8)))  __bf16 v8bf;
typedef __attribute__((ext_vector_type(8)))  float  v8f;
typedef __attribute__((ext_vector_type(4)))  int    v4i;

#define N_ROWS 16384
#define IN_DIM 2048
#define OUT_DIM 1024

// LDS tile geometry: 128 rows x 32 bf16 (64B) padded to 80B per row (16B aligned)
#define ROW_BYTES 80
#define TILE_BYTES (128 * ROW_BYTES)      // 10240
#define BUF_BYTES  (2 * TILE_BYTES)       // A + B per buffer = 20480

#if __has_builtin(__builtin_amdgcn_global_load_async_to_lds_b128) && \
    __has_builtin(__builtin_amdgcn_s_wait_asynccnt)
#define HAS_ASYNC 1
typedef __attribute__((address_space(1))) v4i* gv4ip;   // global int4*
typedef __attribute__((address_space(3))) v4i* lv4ip;   // LDS int4*
#else
#define HAS_ASYNC 0
#endif

// ---------------------------------------------------------------------------
// 1) X (f32) -> Xb (bf16), elementwise, 8 elems/thread.
// ---------------------------------------------------------------------------
__global__ __launch_bounds__(256) void xb_kernel(const float* __restrict__ X,
                                                 __bf16* __restrict__ Xb) {
  const size_t i = ((size_t)blockIdx.x * 256 + threadIdx.x) * 8;
  const float4 a = *(const float4*)(X + i);
  const float4 b = *(const float4*)(X + i + 4);
  v8bf o;
  o[0] = (__bf16)a.x; o[1] = (__bf16)a.y; o[2] = (__bf16)a.z; o[3] = (__bf16)a.w;
  o[4] = (__bf16)b.x; o[5] = (__bf16)b.y; o[6] = (__bf16)b.z; o[7] = (__bf16)b.w;
  *(v8bf*)(Xb + i) = o;
}

// ---------------------------------------------------------------------------
// 2) W (IN x OUT, f32) -> Wt (OUT x IN, bf16) via LDS tile transpose.
// ---------------------------------------------------------------------------
__global__ __launch_bounds__(256) void wt_kernel(const float* __restrict__ W,
                                                 __bf16* __restrict__ Wt) {
  __shared__ float tile[32][33];
  const int k0 = blockIdx.x * 32;
  const int n0 = blockIdx.y * 32;
  const int tx = threadIdx.x;
  const int ty = threadIdx.y;
#pragma unroll
  for (int j = 0; j < 4; ++j) {
    const int k = ty + j * 8;
    tile[k][tx] = W[(size_t)(k0 + k) * OUT_DIM + n0 + tx];
  }
  __syncthreads();
#pragma unroll
  for (int j = 0; j < 4; ++j) {
    const int n = ty + j * 8;
    Wt[(size_t)(n0 + n) * IN_DIM + k0 + tx] = (__bf16)tile[tx][n];
  }
}

// ---------------------------------------------------------------------------
// 3) Zero colsum (ws is poisoned; must re-zero every call).
// ---------------------------------------------------------------------------
__global__ __launch_bounds__(256) void zero_kernel(float* __restrict__ colsum) {
  for (int j = threadIdx.x; j < OUT_DIM; j += 256) colsum[j] = 0.0f;
}

// ---------------------------------------------------------------------------
// 4) GEMM: C = X @ W (bf16 WMMA, f32 acc), LDS double-buffered tiles,
//    async global->LDS staging, K unrolled 2x (static buffer offsets),
//    + column-sum atomics.
//    Block 256 = 8 waves; tile 128(M) x 128(N); wave grid 4(M) x 2(N).
// ---------------------------------------------------------------------------
__global__ __launch_bounds__(256) void gemm_kernel(const __bf16* __restrict__ Xb,
                                                   const __bf16* __restrict__ Wt,
                                                   float* __restrict__ C,
                                                   float* __restrict__ colsum) {
  __shared__ __align__(16) char smem[2 * BUF_BYTES];  // 40 KB
  __shared__ float colAcc[128];

  const int lane  = threadIdx.x & 31;
  const int wave  = threadIdx.x >> 5;
  const int waveM = wave & 3;
  const int waveN = wave >> 2;
  const int tileM = blockIdx.x * 128;   // gridDim.x = 128
  const int tileN = blockIdx.y * 128;   // gridDim.y = 8
  const int half  = lane >> 4;
  const int l15   = lane & 15;

  const int rowBase = tileM + waveM * 32;
  const int colBase = tileN + waveN * 64;

  v8f acc[2][4] = {};

  // ---- hoisted staging descriptors: 4 x 16B chunks per thread per K-step ----
  const __bf16* gbase[4];
  int lbase[4];
#pragma unroll
  for (int j = 0; j < 2; ++j) {            // A chunks (512 chunks / 256 threads)
    const int q = threadIdx.x + j * 256;
    const int r = q >> 2, c = q & 3;
    gbase[j] = Xb + (size_t)(tileM + r) * IN_DIM + c * 8;
    lbase[j] = r * ROW_BYTES + c * 16;
  }
#pragma unroll
  for (int j = 0; j < 2; ++j) {            // B chunks
    const int q = threadIdx.x + j * 256;
    const int r = q >> 2, c = q & 3;
    gbase[j + 2] = Wt + (size_t)(tileN + r) * IN_DIM + c * 8;
    lbase[j + 2] = TILE_BYTES + r * ROW_BYTES + c * 16;
  }

  auto stage = [&](int k0, int bufOff) {
#pragma unroll
    for (int j = 0; j < 4; ++j) {
#if HAS_ASYNC
      __builtin_amdgcn_global_load_async_to_lds_b128(
          (gv4ip)(void*)(gbase[j] + k0),
          (lv4ip)(void*)(smem + bufOff + lbase[j]), 0, 0);
#else
      *(v8bf*)(smem + bufOff + lbase[j]) = *(const v8bf*)(gbase[j] + k0);
#endif
    }
  };

  // ---- hoisted fragment offsets ----
  int offA[2], offB[4];
#pragma unroll
  for (int m = 0; m < 2; ++m)
    offA[m] = (waveM * 32 + m * 16 + l15) * ROW_BYTES + half * 16;
#pragma unroll
  for (int n = 0; n < 4; ++n)
    offB[n] = TILE_BYTES + (waveN * 64 + n * 16 + l15) * ROW_BYTES + half * 16;

  auto compute = [&](int bufOff) {
    // A fragments (16x32 bf16): lanes 0-15 K 0-7|16-23, lanes 16-31 K 8-15|24-31
    v16bf afr[2];
#pragma unroll
    for (int m = 0; m < 2; ++m) {
      const v8bf* pa = (const v8bf*)(smem + bufOff + offA[m]);
      afr[m] = __builtin_shufflevector(pa[0], pa[2], 0, 1, 2, 3, 4, 5, 6, 7,
                                       8, 9, 10, 11, 12, 13, 14, 15);
    }
    // B fragments (32x16 bf16): lane = col; lanes 0-15 K 0-7|16-23, lanes 16-31 K 8-15|24-31
    v16bf bfr[4];
#pragma unroll
    for (int n = 0; n < 4; ++n) {
      const v8bf* pb = (const v8bf*)(smem + bufOff + offB[n]);
      bfr[n] = __builtin_shufflevector(pb[0], pb[2], 0, 1, 2, 3, 4, 5, 6, 7,
                                       8, 9, 10, 11, 12, 13, 14, 15);
    }
#pragma unroll
    for (int m = 0; m < 2; ++m)
#pragma unroll
      for (int n = 0; n < 4; ++n)
        acc[m][n] = __builtin_amdgcn_wmma_f32_16x16x32_bf16(
            false, afr[m], false, bfr[n], (short)0, acc[m][n], false, false);
  };

  // ---- software pipeline: 64-deep K per trip, static buffer offsets ----
  stage(0, 0);
  for (int k0 = 0; k0 < IN_DIM; k0 += 64) {
    stage(k0 + 32, BUF_BYTES);             // k0+32 <= IN_DIM-32 always
#if HAS_ASYNC
    __builtin_amdgcn_s_wait_asynccnt(4);   // buffer-0 group complete
#endif
    __syncthreads();
    compute(0);
    __syncthreads();

    if (k0 + 64 < IN_DIM) {
      stage(k0 + 64, 0);
#if HAS_ASYNC
      __builtin_amdgcn_s_wait_asynccnt(4); // buffer-1 group complete
#endif
    } else {
#if HAS_ASYNC
      __builtin_amdgcn_s_wait_asynccnt(0);
#endif
    }
    __syncthreads();
    compute(BUF_BYTES);
    __syncthreads();
  }

  // ---- store C tile + column partial sums ----
  for (int t = threadIdx.x; t < 128; t += 256) colAcc[t] = 0.0f;
  __syncthreads();
#pragma unroll
  for (int m = 0; m < 2; ++m) {
#pragma unroll
    for (int n = 0; n < 4; ++n) {
      const int col = colBase + n * 16 + l15;
      float partial = 0.0f;
#pragma unroll
      for (int e = 0; e < 8; ++e) {
        const int row = rowBase + m * 16 + half * 8 + e;
        C[(size_t)row * OUT_DIM + col] = acc[m][n][e];
        partial += acc[m][n][e];
      }
      atomicAdd(&colAcc[col - tileN], partial);     // ds_add_f32
    }
  }
  __syncthreads();
  for (int t = threadIdx.x; t < 128; t += 256)
    atomicAdd(&colsum[tileN + t], colAcc[t]);       // global_atomic_add_f32
}

// ---------------------------------------------------------------------------
// 5) scores[i] = sum_j C[i,j] * (a[j]*colsum[j]) ; one wave per row.
// ---------------------------------------------------------------------------
__global__ __launch_bounds__(256) void scores_kernel(const float* __restrict__ C,
                                                     const float* __restrict__ a,
                                                     const float* __restrict__ colsum,
                                                     float* __restrict__ scores) {
  __shared__ float v[OUT_DIM];
  for (int j = threadIdx.x; j < OUT_DIM; j += 256) v[j] = a[j] * colsum[j];
  __syncthreads();

  const int wave = threadIdx.x >> 5;
  const int lane = threadIdx.x & 31;
  const int row  = blockIdx.x * 8 + wave;
  const float* r = C + (size_t)row * OUT_DIM;

  float s = 0.0f;
  for (int j = lane; j < OUT_DIM; j += 32) s += r[j] * v[j];
#pragma unroll
  for (int off = 16; off > 0; off >>= 1) s += __shfl_xor(s, off, 32);
  if (lane == 0) scores[row] = s;
}

// ---------------------------------------------------------------------------
// 6) Softmax stats: stats[0]=max, stats[1]=sum(exp(s-max)).
// ---------------------------------------------------------------------------
__global__ __launch_bounds__(1024) void softmax_stats(const float* __restrict__ scores,
                                                      float* __restrict__ stats) {
  __shared__ float red[1024];
  float m = -INFINITY;
  for (int i = threadIdx.x; i < N_ROWS; i += 1024) m = fmaxf(m, scores[i]);
  red[threadIdx.x] = m;
  __syncthreads();
  for (int off = 512; off > 0; off >>= 1) {
    if (threadIdx.x < off)
      red[threadIdx.x] = fmaxf(red[threadIdx.x], red[threadIdx.x + off]);
    __syncthreads();
  }
  const float mx = red[0];
  __syncthreads();
  float s = 0.0f;
  for (int i = threadIdx.x; i < N_ROWS; i += 1024) s += __expf(scores[i] - mx);
  red[threadIdx.x] = s;
  __syncthreads();
  for (int off = 512; off > 0; off >>= 1) {
    if (threadIdx.x < off) red[threadIdx.x] += red[threadIdx.x + off];
    __syncthreads();
  }
  if (threadIdx.x == 0) { stats[0] = mx; stats[1] = red[0]; }
}

// ---------------------------------------------------------------------------
// 7) In-place row scale: out[i,:] *= exp(scores[i]-max)/sumexp.
// ---------------------------------------------------------------------------
__global__ __launch_bounds__(256) void scale_kernel(const float* __restrict__ scores,
                                                    const float* __restrict__ stats,
                                                    float* __restrict__ out) {
  const size_t base = ((size_t)blockIdx.x * 256 + threadIdx.x) * 4;
  const int row = (int)(base >> 10);
  const float w = __expf(scores[row] - stats[0]) / stats[1];
  float4 c = *(const float4*)(out + base);
  c.x *= w; c.y *= w; c.z *= w; c.w *= w;
  *(float4*)(out + base) = c;
}

// ---------------------------------------------------------------------------
extern "C" void kernel_launch(void* const* d_in, const int* in_sizes, int n_in,
                              void* d_out, int out_size, void* d_ws, size_t ws_size,
                              hipStream_t stream) {
  const float* X = (const float*)d_in[0];   // 16384 x 2048
  const float* W = (const float*)d_in[1];   // 2048 x 1024
  const float* a = (const float*)d_in[2];   // 1024
  float* C = (float*)d_out;                 // w_input lives in d_out; scaled in place

  // workspace: Xb (64MB bf16) | Wt (4MB bf16) | colsum | scores | stats
  char* ws = (char*)d_ws;
  __bf16* Xb     = (__bf16*)ws;
  __bf16* Wt     = (__bf16*)(ws + (size_t)N_ROWS * IN_DIM * 2);
  float*  colsum = (float*)(ws + (size_t)N_ROWS * IN_DIM * 2
                               + (size_t)IN_DIM * OUT_DIM * 2);
  float*  scores = colsum + OUT_DIM;
  float*  stats  = scores + N_ROWS;

  xb_kernel<<<(N_ROWS * (size_t)IN_DIM) / (256 * 8), 256, 0, stream>>>(X, Xb);
  wt_kernel<<<dim3(IN_DIM / 32, OUT_DIM / 32), dim3(32, 8), 0, stream>>>(W, Wt);
  zero_kernel<<<1, 256, 0, stream>>>(colsum);
  gemm_kernel<<<dim3(N_ROWS / 128, OUT_DIM / 128), 256, 0, stream>>>(Xb, Wt, C, colsum);
  scores_kernel<<<N_ROWS / 8, 256, 0, stream>>>(C, a, colsum, scores);
  softmax_stats<<<1, 1024, 0, stream>>>(scores, stats);
  scale_kernel<<<(N_ROWS * OUT_DIM) / (256 * 4), 256, 0, stream>>>(scores, stats, C);
}
